// GraphConv_48455821033981
// MI455X (gfx1250) — matrix operand order repaired and checked by reading, not compiled, not run on hardware
//
#include <hip/hip_runtime.h>
#include <hip/hip_bf16.h>

typedef __attribute__((ext_vector_type(16))) __bf16 v16bf;
typedef __attribute__((ext_vector_type(8)))  __bf16 bf16x8;
typedef __attribute__((ext_vector_type(4)))  __bf16 bf16x4;
typedef __attribute__((ext_vector_type(8)))  float  v8f;

#define NNODES 50000
#define NEDGES 800000
#define FEAT   128      // D == H == 128
#define NGRAPH 128
#define NCLASS 10
#define SCANB  256      // scan block size
#define NSCANB ((NNODES + SCANB - 1) / SCANB)   // 196 (<= 256, single-level rescan)

// ---------------------------------------------------------------- zero fill
__global__ void zero_u32_kernel(unsigned* __restrict__ p, long n) {
    long i = (long)blockIdx.x * blockDim.x + threadIdx.x;
    if (i < n) p[i] = 0u;
}

// -------------------------------------------------------- f32 -> bf16 convert
__global__ void cvt_bf16_kernel(const float* __restrict__ x, __bf16* __restrict__ xb, long n4) {
    long i = (long)blockIdx.x * blockDim.x + threadIdx.x;
    if (i >= n4) return;
    const float4 v = *(const float4*)(x + i * 4);
    bf16x4 o;
    o[0] = (__bf16)v.x; o[1] = (__bf16)v.y; o[2] = (__bf16)v.z; o[3] = (__bf16)v.w;
    *(bf16x4*)(xb + i * 4) = o;
}

// ------------------------------------------------------------- CSR build
__global__ void degree_kernel(const int* __restrict__ dst, int* __restrict__ deg, int nedges) {
    int e = blockIdx.x * blockDim.x + threadIdx.x;
    if (e < nedges) atomicAdd(&deg[dst[e]], 1);
}

// exclusive scan, stage 1: per-block scan (Hillis-Steele), write partials + block sums
__global__ __launch_bounds__(SCANB) void scan1_kernel(
    const int* __restrict__ deg, int* __restrict__ part, int* __restrict__ bsum, int n)
{
    __shared__ int tmp[SCANB];
    int i = blockIdx.x * SCANB + threadIdx.x;
    int v = (i < n) ? deg[i] : 0;
    tmp[threadIdx.x] = v;
    __syncthreads();
    for (int off = 1; off < SCANB; off <<= 1) {
        int t = (threadIdx.x >= off) ? tmp[threadIdx.x - off] : 0;
        __syncthreads();
        tmp[threadIdx.x] += t;
        __syncthreads();
    }
    if (i < n) part[i] = tmp[threadIdx.x] - v;              // exclusive
    if (threadIdx.x == SCANB - 1) bsum[blockIdx.x] = tmp[SCANB - 1];
}

// stage 2: single-block exclusive scan of block sums (NSCANB <= 256)
__global__ __launch_bounds__(SCANB) void scan2_kernel(int* __restrict__ bsum, int nb) {
    __shared__ int tmp[SCANB];
    int v = (threadIdx.x < nb) ? bsum[threadIdx.x] : 0;
    tmp[threadIdx.x] = v;
    __syncthreads();
    for (int off = 1; off < SCANB; off <<= 1) {
        int t = (threadIdx.x >= off) ? tmp[threadIdx.x - off] : 0;
        __syncthreads();
        tmp[threadIdx.x] += t;
        __syncthreads();
    }
    if (threadIdx.x < nb) bsum[threadIdx.x] = tmp[threadIdx.x] - v;  // exclusive
}

// stage 3: add block offsets -> rowStart, duplicate into cursor, cap with E
__global__ __launch_bounds__(SCANB) void scan3_kernel(
    int* __restrict__ rowStart, int* __restrict__ cursor,
    const int* __restrict__ bsum, int n)
{
    int i = blockIdx.x * SCANB + threadIdx.x;
    if (i < n) {
        int v = rowStart[i] + bsum[blockIdx.x];
        rowStart[i] = v;
        cursor[i]   = v;
    }
    if (i == 0) rowStart[n] = NEDGES;
}

__global__ void fill_csr_kernel(const int* __restrict__ src, const int* __restrict__ dst,
                                int* __restrict__ cursor, int* __restrict__ csrc, int nedges) {
    int e = blockIdx.x * blockDim.x + threadIdx.x;
    if (e >= nedges) return;
    int pos = atomicAdd(&cursor[dst[e]], 1);
    csrc[pos] = src[e];
}

// canonical (ascending-src) order per row -> deterministic float sums
__global__ void sort_rows_kernel(const int* __restrict__ rowStart, int* __restrict__ csrc, int n) {
    int node = blockIdx.x * blockDim.x + threadIdx.x;
    if (node >= n) return;
    int s0 = rowStart[node], s1 = rowStart[node + 1];
    for (int i = s0 + 1; i < s1; ++i) {
        int v = csrc[i], j = i - 1;
        while (j >= s0 && csrc[j] > v) { csrc[j + 1] = csrc[j]; --j; }
        csrc[j + 1] = v;
    }
}

// ------------------------------------------- gather: agg[n] = sum_{s in N(n)} h[s]
// one wave per node; lane owns 4 features. Indices are fetched cooperatively
// (one coalesced b32 per 32 neighbors) and broadcast with __shfl in ascending
// order, keeping the canonical deterministic summation order.
__global__ __launch_bounds__(256) void gather_kernel(
    const __bf16* __restrict__ h, const int* __restrict__ rowStart,
    const int* __restrict__ csrc, __bf16* __restrict__ agg, int nnodes)
{
    long t = (long)blockIdx.x * blockDim.x + threadIdx.x;
    int node = (int)(t >> 5);
    int lane = (int)(t & 31);
    if (node >= nnodes) return;
    int s0 = rowStart[node], s1 = rowStart[node + 1];
    float a0 = 0.f, a1 = 0.f, a2 = 0.f, a3 = 0.f;
    for (int base = s0; base < s1; base += 32) {
        int m = s1 - base;
        int cnt = m < 32 ? m : 32;
        int idx = (lane < cnt) ? csrc[base + lane] : 0;
        for (int j = 0; j < cnt; ++j) {
            int s = __shfl(idx, j, 32);
            bf16x4 v = *(const bf16x4*)(h + (size_t)s * FEAT + lane * 4);
            a0 += (float)v[0]; a1 += (float)v[1]; a2 += (float)v[2]; a3 += (float)v[3];
        }
    }
    bf16x4 o;
    o[0] = (__bf16)a0; o[1] = (__bf16)a1; o[2] = (__bf16)a2; o[3] = (__bf16)a3;
    *(bf16x4*)(agg + (size_t)node * FEAT + lane * 4) = o;
}

// ----------------------------------------------------------------- WMMA GEMM
// hout = relu(agg @ Wr^T + br + hin @ Wroot^T); activations + weights bf16, f32 accum
// Weights are pre-converted to bf16 in global -> staging is pure b128 copies.
__global__ __launch_bounds__(256) void gemm_bias_relu_kernel(
    const __bf16* __restrict__ agg, const __bf16* __restrict__ hin,
    const __bf16* __restrict__ Wrb, const __bf16* __restrict__ Wtb,
    const float* __restrict__ br,  __bf16* __restrict__ hout, int nrows)
{
    __shared__ __align__(16) __bf16 sWr[FEAT * FEAT];   // row-major [n][k], bf16
    __shared__ __align__(16) __bf16 sWt[FEAT * FEAT];
    {
        const uint4* gr = (const uint4*)Wrb;
        const uint4* gt = (const uint4*)Wtb;
        uint4* lr = (uint4*)sWr;
        uint4* lt = (uint4*)sWt;
        for (int i = threadIdx.x; i < (FEAT * FEAT) / 8; i += 256) {   // 2048 x 16B
            lr[i] = gr[i];
            lt[i] = gt[i];
        }
    }
    __syncthreads();

    const int wave = threadIdx.x >> 5;
    const int lane = threadIdx.x & 31;
    const int half = lane >> 4;
    const int nl   = lane & 15;
    const int rowBase = blockIdx.x * 128 + wave * 16;
    const int arow    = rowBase + nl;
    const bool ok     = (arow < nrows);
    const __bf16* aggRow = agg + (size_t)arow * FEAT;
    const __bf16* hRow   = hin + (size_t)arow * FEAT;

    v8f acc[8];
#pragma unroll
    for (int ct = 0; ct < 8; ++ct) { v8f z = {}; acc[ct] = z; }

#pragma unroll
    for (int kb = 0; kb < 4; ++kb) {
        // A 16x32 bf16 layout: lane-half h holds K = {8h..8h+7} U {16+8h..16+8h+7}
        v16bf aAgg, aH;
        if (ok) {
            bf16x8 lo = *(const bf16x8*)(aggRow + kb * 32 + 8 * half);
            bf16x8 hi = *(const bf16x8*)(aggRow + kb * 32 + 16 + 8 * half);
            aAgg = __builtin_shufflevector(lo, hi, 0,1,2,3,4,5,6,7,8,9,10,11,12,13,14,15);
            lo = *(const bf16x8*)(hRow + kb * 32 + 8 * half);
            hi = *(const bf16x8*)(hRow + kb * 32 + 16 + 8 * half);
            aH = __builtin_shufflevector(lo, hi, 0,1,2,3,4,5,6,7,8,9,10,11,12,13,14,15);
        } else {
            v16bf z = {}; aAgg = z; aH = z;
        }
#pragma unroll
        for (int ct = 0; ct < 8; ++ct) {
            const int n = ct * 16 + nl;
            v16bf bR, bT;
#pragma unroll
            for (int v = 0; v < 8; ++v) {
                int k0 = kb * 32 + 8 * half + 2 * v + 8 * (v >> 2);   // B: lane-half h -> K = {8h..8h+7} U {16+8h..16+8h+7}
                const __bf16* pr = &sWr[n * FEAT + k0];
                const __bf16* pt = &sWt[n * FEAT + k0];
                bR[2 * v] = pr[0]; bR[2 * v + 1] = pr[1];
                bT[2 * v] = pt[0]; bT[2 * v + 1] = pt[1];
            }
            acc[ct] = __builtin_amdgcn_wmma_f32_16x16x32_bf16(
                false, aAgg, false, bR, (short)0, acc[ct], false, false);
            acc[ct] = __builtin_amdgcn_wmma_f32_16x16x32_bf16(
                false, aH,   false, bT, (short)0, acc[ct], false, false);
        }
    }

    // epilogue: bias + relu, store bf16.  C layout: VGPR r -> M = r + 8*half
#pragma unroll
    for (int ct = 0; ct < 8; ++ct) {
        const int n = ct * 16 + nl;
        const float bias = br[n];
#pragma unroll
        for (int r = 0; r < 8; ++r) {
            int row = rowBase + r + 8 * half;
            if (row < nrows) {
                float val = acc[ct][r] + bias;
                hout[(size_t)row * FEAT + n] = (__bf16)(val > 0.f ? val : 0.f);
            }
        }
    }
}

// ----------------------------------------- graph ranges (batch is sorted)
__global__ void graph_ranges_kernel(const int* __restrict__ batch, int* __restrict__ gstart) {
    int g = blockIdx.x * blockDim.x + threadIdx.x;
    if (g > NGRAPH) return;
    if (g == NGRAPH) { gstart[g] = NNODES; return; }
    int lo = 0, hi = NNODES;
    while (lo < hi) { int mid = (lo + hi) >> 1; if (batch[mid] < g) lo = mid + 1; else hi = mid; }
    gstart[g] = lo;
}

// ------------------- deterministic mean-pool: one block (8 waves) per graph
__global__ __launch_bounds__(256) void pool_kernel(
    const __bf16* __restrict__ h, const int* __restrict__ gstart, float* __restrict__ emb)
{
    __shared__ float part[8][FEAT];
    const int g    = blockIdx.x;
    const int wave = threadIdx.x >> 5;
    const int lane = threadIdx.x & 31;
    const int s0 = gstart[g], s1 = gstart[g + 1];
    float a0 = 0.f, a1 = 0.f, a2 = 0.f, a3 = 0.f;
    for (int i = s0 + wave; i < s1; i += 8) {
        bf16x4 v = *(const bf16x4*)(h + (size_t)i * FEAT + lane * 4);
        a0 += (float)v[0]; a1 += (float)v[1]; a2 += (float)v[2]; a3 += (float)v[3];
    }
    part[wave][lane * 4 + 0] = a0;
    part[wave][lane * 4 + 1] = a1;
    part[wave][lane * 4 + 2] = a2;
    part[wave][lane * 4 + 3] = a3;
    __syncthreads();
    const int f = threadIdx.x;
    if (f < FEAT) {
        float s = 0.f;
#pragma unroll
        for (int k = 0; k < 8; ++k) s += part[k][f];
        int cnt = s1 - s0; if (cnt < 1) cnt = 1;
        emb[(size_t)g * FEAT + f] = s / (float)cnt;
    }
}

// --------------------------------------------------------------- classifier
__global__ void classify_kernel(
    const float* __restrict__ emb, const float* __restrict__ lin_w,
    const float* __restrict__ lin_b, float* __restrict__ out)
{
    int idx = blockIdx.x * blockDim.x + threadIdx.x;
    if (idx >= NGRAPH * NCLASS) return;
    int g = idx / NCLASS;
    int c = idx % NCLASS;
    float s = lin_b[c];
    const float* e = emb + (size_t)g * FEAT;
    const float* w = lin_w + (size_t)c * FEAT;
#pragma unroll 8
    for (int k = 0; k < FEAT; ++k) s += e[k] * w[k];
    out[idx] = s;
}

extern "C" void kernel_launch(void* const* d_in, const int* in_sizes, int n_in,
                              void* d_out, int out_size, void* d_ws, size_t ws_size,
                              hipStream_t stream) {
    const float* x        = (const float*)d_in[0];
    const int*   edge     = (const int*)d_in[1];      // [2][E]
    const int*   batch    = (const int*)d_in[2];
    const float* Wr[4]    = { (const float*)d_in[3],  (const float*)d_in[6],
                              (const float*)d_in[9],  (const float*)d_in[12] };
    const float* br[4]    = { (const float*)d_in[4],  (const float*)d_in[7],
                              (const float*)d_in[10], (const float*)d_in[13] };
    const float* Wroot[4] = { (const float*)d_in[5],  (const float*)d_in[8],
                              (const float*)d_in[11], (const float*)d_in[14] };
    const float* lin_w    = (const float*)d_in[15];
    const float* lin_b    = (const float*)d_in[16];
    float*       out      = (float*)d_out;

    const int* src = edge;
    const int* dst = edge + NEDGES;

    // ---- workspace layout
    const long NH = (long)NNODES * FEAT;              // 6,400,000 elements
    const long WH = (long)FEAT * FEAT;                // 16,384 elements
    char* ws = (char*)d_ws;
    __bf16* xbf   = (__bf16*)ws;            ws += NH * 2;
    __bf16* hA    = (__bf16*)ws;            ws += NH * 2;
    __bf16* hB    = (__bf16*)ws;            ws += NH * 2;
    __bf16* aggb  = (__bf16*)ws;            ws += NH * 2;
    __bf16* wrb   = (__bf16*)ws;            ws += WH * 4 * 2;   // 4 layers Wr
    __bf16* wtb   = (__bf16*)ws;            ws += WH * 4 * 2;   // 4 layers Wroot
    int* csrc     = (int*)ws;               ws += (long)NEDGES * 4;
    int* deg      = (int*)ws;               ws += (long)NNODES * 4;
    int* rowStart = (int*)ws;               ws += (long)(NNODES + 1) * 4;
    int* cursor   = (int*)ws;               ws += (long)NNODES * 4;
    int* bsum     = (int*)ws;               ws += SCANB * 4;
    int* gstart   = (int*)ws;               ws += (NGRAPH + 1) * 4;
    float* emb    = (float*)ws;             ws += (long)NGRAPH * FEAT * 4;

    // ---- one-time per launch: bf16 features/weights, CSR, graph ranges
    cvt_bf16_kernel<<<(int)((NH / 4 + 255) / 256), 256, 0, stream>>>(x, xbf, NH / 4);
    for (int l = 0; l < 4; ++l) {
        cvt_bf16_kernel<<<(int)(WH / 4 / 256), 256, 0, stream>>>(Wr[l],    wrb + l * WH, WH / 4);
        cvt_bf16_kernel<<<(int)(WH / 4 / 256), 256, 0, stream>>>(Wroot[l], wtb + l * WH, WH / 4);
    }

    zero_u32_kernel<<<(NNODES + 255) / 256, 256, 0, stream>>>((unsigned*)deg, NNODES);
    degree_kernel<<<(NEDGES + 255) / 256, 256, 0, stream>>>(dst, deg, NEDGES);
    scan1_kernel<<<NSCANB, SCANB, 0, stream>>>(deg, rowStart, bsum, NNODES);
    scan2_kernel<<<1, SCANB, 0, stream>>>(bsum, NSCANB);
    scan3_kernel<<<NSCANB, SCANB, 0, stream>>>(rowStart, cursor, bsum, NNODES);
    fill_csr_kernel<<<(NEDGES + 255) / 256, 256, 0, stream>>>(src, dst, cursor, csrc, NEDGES);
    sort_rows_kernel<<<(NNODES + 255) / 256, 256, 0, stream>>>(rowStart, csrc, NNODES);
    graph_ranges_kernel<<<1, 256, 0, stream>>>(batch, gstart);

    // ---- 4 GraphConv layers
    const long gatherT   = (long)NNODES * 32;
    const int  gatherBlk = (int)((gatherT + 255) / 256);
    const int  gemmBlk   = (NNODES + 127) / 128;      // 391

    const __bf16* h_in  = xbf;
    __bf16*       h_out = hA;
    for (int l = 0; l < 4; ++l) {
        gather_kernel<<<gatherBlk, 256, 0, stream>>>(h_in, rowStart, csrc, aggb, NNODES);
        gemm_bias_relu_kernel<<<gemmBlk, 256, 0, stream>>>(
            aggb, h_in, wrb + l * WH, wtb + l * WH, br[l], h_out, NNODES);
        h_in  = h_out;
        h_out = (h_out == hA) ? hB : hA;
    }

    // ---- pool + classify
    pool_kernel<<<NGRAPH, 256, 0, stream>>>(h_in, gstart, emb);
    classify_kernel<<<(NGRAPH * NCLASS + 255) / 256, 256, 0, stream>>>(emb, lin_w, lin_b, out);
}
